// MixtralForCausalLM_87462714016326
// MI455X (gfx1250) — matrix operand order repaired and checked by reading, not compiled, not run on hardware
//
#include <hip/hip_runtime.h>
#include <hip/hip_bf16.h>
#include <stdint.h>

#define NUM_EXPERTS 8
#define TOP_K 2
#define HIDDEN 1024
#define FFN 3584
#define TOKENS 16384

#define BM 128
#define BN 128
#define BK 32
#define BKP 40  // padded LDS row (elems); 80B rows keep 16B alignment, spread banks

typedef __attribute__((ext_vector_type(16))) __bf16 v16bf;
typedef __attribute__((ext_vector_type(8)))  __bf16 bf16x8;
typedef __attribute__((ext_vector_type(8)))  float  v8f;

typedef __attribute__((ext_vector_type(4))) unsigned int u32x4;
typedef __attribute__((ext_vector_type(8))) int          i32x8;
typedef __attribute__((ext_vector_type(4))) int          i32x4;

#if __has_builtin(__builtin_amdgcn_tensor_load_to_lds) && \
    __has_builtin(__builtin_amdgcn_s_wait_tensorcnt)
#define USE_TDM 1
#else
#define USE_TDM 0
#endif

__device__ __forceinline__ unsigned short f32_to_bf16_rne(float f) {
  unsigned int u = __float_as_uint(f);
  u += 0x7FFFu + ((u >> 16) & 1u);  // round-to-nearest-even
  return (unsigned short)(u >> 16);
}

// Build a v16bf WMMA fragment from two 16-byte LDS chunks (2x ds_load_b128).
__device__ __forceinline__ v16bf frag2(const unsigned short* p0, const unsigned short* p1) {
  bf16x8 lo = *(const bf16x8*)p0;
  bf16x8 hi = *(const bf16x8*)p1;
  v16bf r;
#pragma unroll
  for (int i = 0; i < 8; ++i) { r[i] = lo[i]; r[i + 8] = hi[i]; }
  return r;
}

// ---------------- small utility kernels ----------------

__global__ void zero_f32_kernel(float* p, int n4) {
  int i = blockIdx.x * blockDim.x + threadIdx.x;
  if (i < n4) ((float4*)p)[i] = make_float4(0.f, 0.f, 0.f, 0.f);
}

__global__ void zero_counts_kernel(int* counts) {
  if (threadIdx.x < NUM_EXPERTS) counts[threadIdx.x] = 0;
}

__global__ void offsets_kernel(const int* counts, int* offsets) {
  if (threadIdx.x == 0) {
    int acc = 0;
    for (int e = 0; e < NUM_EXPERTS; ++e) { offsets[e] = acc; acc += counts[e]; }
    offsets[NUM_EXPERTS] = acc;
  }
}

__global__ void scatter_kernel(const int* __restrict__ tokExp, const int* __restrict__ tokPos,
                               const float* __restrict__ tokW, const int* __restrict__ offsets,
                               int* __restrict__ entTok, float* __restrict__ entW) {
  int i = blockIdx.x * blockDim.x + threadIdx.x;
  if (i < TOP_K * TOKENS) {
    int e = tokExp[i];
    int idx = offsets[e] + tokPos[i];
    entTok[idx] = i >> 1;
    entW[idx] = tokW[i];
  }
}

// ---------------- router: logits -> softmax -> top-2 -> per-expert lists ----------------

__global__ void __launch_bounds__(256)
router_kernel(const float* __restrict__ X, const float* __restrict__ G,
              int* counts, int* tokExp, int* tokPos, float* tokW) {
  __shared__ float gw[HIDDEN * NUM_EXPERTS];  // 32 KB staged gate weights
  for (int i = threadIdx.x; i < HIDDEN * NUM_EXPERTS; i += blockDim.x) gw[i] = G[i];
  __syncthreads();

  const int t = blockIdx.x * blockDim.x + threadIdx.x;
  float acc[NUM_EXPERTS];
#pragma unroll
  for (int e = 0; e < NUM_EXPERTS; ++e) acc[e] = 0.f;

  const float4* xrow = (const float4*)(X + (size_t)t * HIDDEN);
  for (int k4 = 0; k4 < HIDDEN / 4; ++k4) {
    float4 x = xrow[k4];
    const float* g0 = &gw[(k4 * 4) * NUM_EXPERTS];
#pragma unroll
    for (int e = 0; e < NUM_EXPERTS; ++e)
      acc[e] += x.x * g0[e] + x.y * g0[NUM_EXPERTS + e] +
                x.z * g0[2 * NUM_EXPERTS + e] + x.w * g0[3 * NUM_EXPERTS + e];
  }

  float mx = acc[0];
#pragma unroll
  for (int e = 1; e < NUM_EXPERTS; ++e) mx = fmaxf(mx, acc[e]);
  float p[NUM_EXPERTS];
#pragma unroll
  for (int e = 0; e < NUM_EXPERTS; ++e) p[e] = __expf(acc[e] - mx);

  int i0 = 0;
#pragma unroll
  for (int e = 1; e < NUM_EXPERTS; ++e) if (p[e] > p[i0]) i0 = e;
  int i1 = (i0 == 0) ? 1 : 0;
#pragma unroll
  for (int e = 0; e < NUM_EXPERTS; ++e) if (e != i0 && p[e] > p[i1]) i1 = e;

  float denom = p[i0] + p[i1];
  float wa = p[i0] / denom;
  float wb = p[i1] / denom;

#pragma unroll
  for (int s = 0; s < TOP_K; ++s) {
    int e = s ? i1 : i0;
    float w = s ? wb : wa;
    int pos = atomicAdd(&counts[e], 1);
    tokExp[t * TOP_K + s] = e;
    tokPos[t * TOP_K + s] = pos;
    tokW[t * TOP_K + s] = w;
  }
}

// ---------------- GEMM1: H = silu(X@w1) * (X@w3), gathered rows, bf16 WMMA ----------------

__global__ void __launch_bounds__(256)
moe_gemm1_kernel(const float* __restrict__ X, const float* __restrict__ W1,
                 const float* __restrict__ W3, const int* __restrict__ counts,
                 const int* __restrict__ offsets, const int* __restrict__ entTok,
                 unsigned short* __restrict__ Hbuf) {
  const int e = blockIdx.z;
  const int cnt = counts[e];
  const int mTile = blockIdx.y;
  if (mTile * BM >= cnt) return;
  const int nTile = blockIdx.x;
  const int base = offsets[e];

  __shared__ unsigned short ldsA[BM * BKP];
  __shared__ unsigned short ldsB1[BN * BKP];
  __shared__ unsigned short ldsB3[BN * BKP];
  __shared__ int ldsTok[BM];

  const int tid = threadIdx.x;
  if (tid < BM) {
    int m = mTile * BM + tid;
    ldsTok[tid] = (m < cnt) ? entTok[base + m] : -1;
  }
  __syncthreads();

  const int lane = tid & 31;
  const int wave = tid >> 5;
  const int wm = wave & 3;   // 4 waves along M (32 rows each)
  const int wn = wave >> 2;  // 2 waves along N (64 cols each)

  v8f accG[2][4], accU[2][4];
#pragma unroll
  for (int i = 0; i < 2; ++i)
#pragma unroll
    for (int j = 0; j < 4; ++j) { accG[i][j] = (v8f)(0.f); accU[i][j] = (v8f)(0.f); }

  const float* w1base = W1 + (size_t)e * HIDDEN * FFN + (size_t)nTile * BN;
  const float* w3base = W3 + (size_t)e * HIDDEN * FFN + (size_t)nTile * BN;

  const int arow = tid & (BM - 1);
  const int acol0 = (tid >> 7) * 16;  // 0 or 16
  const int brow = tid & (BK - 1);
  const int bcol0 = (tid >> 5) * 16;  // 0..112

  for (int k0 = 0; k0 < HIDDEN; k0 += BK) {
    // ---- fill A tile (gathered token rows, f32 -> bf16) ----
    {
      int tok = ldsTok[arow];
      const float* src = X + (size_t)tok * HIDDEN + k0 + acol0;
#pragma unroll
      for (int j = 0; j < 4; ++j) {
        float4 v = (tok >= 0) ? *(const float4*)(src + j * 4) : make_float4(0.f, 0.f, 0.f, 0.f);
        ushort4 d;
        d.x = f32_to_bf16_rne(v.x); d.y = f32_to_bf16_rne(v.y);
        d.z = f32_to_bf16_rne(v.z); d.w = f32_to_bf16_rne(v.w);
        *(ushort4*)&ldsA[arow * BKP + acol0 + j * 4] = d;
      }
    }
    // ---- fill B tiles transposed ([n][k]) for w1 and w3 ----
    {
      const float* s1 = w1base + (size_t)(k0 + brow) * FFN + bcol0;
      const float* s3 = w3base + (size_t)(k0 + brow) * FFN + bcol0;
      __builtin_prefetch(s1 + (size_t)BK * FFN, 0, 1);
      __builtin_prefetch(s3 + (size_t)BK * FFN, 0, 1);
#pragma unroll
      for (int j = 0; j < 4; ++j) {
        float4 v = *(const float4*)(s1 + j * 4);
        int n = bcol0 + j * 4;
        ldsB1[(n + 0) * BKP + brow] = f32_to_bf16_rne(v.x);
        ldsB1[(n + 1) * BKP + brow] = f32_to_bf16_rne(v.y);
        ldsB1[(n + 2) * BKP + brow] = f32_to_bf16_rne(v.z);
        ldsB1[(n + 3) * BKP + brow] = f32_to_bf16_rne(v.w);
        float4 u = *(const float4*)(s3 + j * 4);
        ldsB3[(n + 0) * BKP + brow] = f32_to_bf16_rne(u.x);
        ldsB3[(n + 1) * BKP + brow] = f32_to_bf16_rne(u.y);
        ldsB3[(n + 2) * BKP + brow] = f32_to_bf16_rne(u.z);
        ldsB3[(n + 3) * BKP + brow] = f32_to_bf16_rne(u.w);
      }
    }
    __syncthreads();

    // ---- WMMA compute: 2 M-frags x 4 N-frags x {w1,w3} ----
    v16bf afrag[2];
#pragma unroll
    for (int i = 0; i < 2; ++i) {
      int m = wm * 32 + i * 16 + (lane & 15);
      int kb = (lane >> 4) * 8;  // A layout: K chunks at kb and kb+16
      afrag[i] = frag2(&ldsA[m * BKP + kb], &ldsA[m * BKP + kb + 16]);
    }
#pragma unroll
    for (int j = 0; j < 4; ++j) {
      int n = wn * 64 + j * 16 + (lane & 15);
      int kb = (lane >> 4) * 8;  // B layout: K chunks at kb and kb+16
      v16bf b1 = frag2(&ldsB1[n * BKP + kb], &ldsB1[n * BKP + kb + 16]);
      v16bf b3 = frag2(&ldsB3[n * BKP + kb], &ldsB3[n * BKP + kb + 16]);
#pragma unroll
      for (int i = 0; i < 2; ++i) {
        accG[i][j] = __builtin_amdgcn_wmma_f32_16x16x32_bf16(
            false, afrag[i], false, b1, (short)0, accG[i][j], false, false);
        accU[i][j] = __builtin_amdgcn_wmma_f32_16x16x32_bf16(
            false, afrag[i], false, b3, (short)0, accU[i][j], false, false);
      }
    }
    __syncthreads();
  }

  // ---- epilogue: SwiGLU, write bf16 H ----
  const int mstart = mTile * BM;
#pragma unroll
  for (int i = 0; i < 2; ++i) {
#pragma unroll
    for (int j = 0; j < 4; ++j) {
      int col = nTile * BN + wn * 64 + j * 16 + (lane & 15);
      int rbase = wm * 32 + i * 16 + (lane >> 4) * 8;
#pragma unroll
      for (int r = 0; r < 8; ++r) {
        int m = mstart + rbase + r;
        if (m < cnt) {
          float g = accG[i][j][r];
          float u = accU[i][j][r];
          float h = (g / (1.0f + __expf(-g))) * u;  // silu(g) * u
          Hbuf[(size_t)(base + m) * FFN + col] = f32_to_bf16_rne(h);
        }
      }
    }
  }
}

// ---------------- GEMM2: Out += combine_w * (H @ w2), bf16 WMMA ----------------
// A tile (bf16 H rows) is moved by the Tensor Data Mover: a 2D D# with
// data_size=2B, tile 32x128, row stride FFN, and hardware LDS padding
// (pad_interval=16 DW = 64B rows, pad_amount=4 DW = 16B) reproducing the
// BKP=40 padded layout the WMMA fragment loaders expect. tensor_dim1 is the
// number of valid rows so OOB rows are zero-filled by the TDM.

__global__ void __launch_bounds__(256)
moe_gemm2_kernel(const unsigned short* __restrict__ Hbuf, const float* __restrict__ W2,
                 const int* __restrict__ counts, const int* __restrict__ offsets,
                 const int* __restrict__ entTok, const float* __restrict__ entW,
                 float* __restrict__ Out) {
  const int e = blockIdx.z;
  const int cnt = counts[e];
  const int mTile = blockIdx.y;
  if (mTile * BM >= cnt) return;
  const int nTile = blockIdx.x;  // HIDDEN/BN = 8
  const int base = offsets[e];

  __shared__ unsigned short ldsA[BM * BKP];
  __shared__ unsigned short ldsB[BN * BKP];
  __shared__ int ldsTok[BM];
  __shared__ float ldsW[BM];

  const int tid = threadIdx.x;
  if (tid < BM) {
    int m = mTile * BM + tid;
    ldsTok[tid] = (m < cnt) ? entTok[base + m] : 0;
    ldsW[tid] = (m < cnt) ? entW[base + m] : 0.f;
  }
  __syncthreads();

  const int lane = tid & 31;
  const int wave = tid >> 5;
  const int wm = wave & 3;
  const int wn = wave >> 2;

  v8f acc[2][4];
#pragma unroll
  for (int i = 0; i < 2; ++i)
#pragma unroll
    for (int j = 0; j < 4; ++j) acc[i][j] = (v8f)(0.f);

  const float* w2base = W2 + (size_t)e * FFN * HIDDEN + (size_t)nTile * BN;

  const int brow = tid & (BK - 1);
  const int bcol0 = (tid >> 5) * 16;

#if USE_TDM
  // Descriptor pieces that do not change across the K loop.
  int rows = cnt - mTile * BM;
  if (rows > BM) rows = BM;
  const unsigned int ldsA_off = (unsigned int)(uintptr_t)(&ldsA[0]);
  // D# group 1: wg_mask=0 | data_size=1(2B) | pad_enable | pad_interval=3(16DW) |
  //             pad_amount=3(4DW); tensor_dim0=BK, tensor_dim1=rows,
  //             tile_dim0=BK, tile_dim1=BM, stride0=FFN.
  i32x8 g1;
  g1[0] = (int)((1u << 16) | (1u << 20) | (3u << 22) | (3u << 25));
  g1[1] = (int)(((unsigned)BK & 0xFFFFu) << 16);
  g1[2] = (int)(((unsigned)rows & 0xFFFFu) << 16);
  g1[3] = (int)(((unsigned)BK) << 16);
  g1[4] = (int)((unsigned)BM);
  g1[5] = (int)((unsigned)FFN);  // tensor_dim0_stride (elements)
  g1[6] = 0;
  g1[7] = 0;
  const i32x4 gz4 = (i32x4)(0);
  const i32x8 gz8 = (i32x8)(0);
  const unsigned long long rowBaseAddr =
      (unsigned long long)(uintptr_t)(Hbuf + (size_t)(base + mTile * BM) * FFN);
#else
  const int arow = tid & (BM - 1);
  const int acol0 = (tid >> 7) * 16;
  const int am = mTile * BM + arow;
  const bool avalid = (am < cnt);
#endif

  for (int k0 = 0; k0 < FFN; k0 += BK) {
#if USE_TDM
    // ---- issue TDM for the A tile (one wave; TDM ignores EXEC) ----
    if (tid == 0) {
      unsigned long long gaddr = rowBaseAddr + (unsigned long long)(k0 * 2);
      u32x4 g0;
      g0[0] = 1u;                                   // count=1 valid descriptor
      g0[1] = ldsA_off;                             // lds_addr (bytes)
      g0[2] = (unsigned int)(gaddr & 0xFFFFFFFFull);
      g0[3] = (unsigned int)((gaddr >> 32) & 0x01FFFFFFull) | (2u << 30);  // type=2
      __builtin_amdgcn_tensor_load_to_lds(g0, g1, gz4, gz4, gz8, 0);
    }
#else
    {
      uint4 v0 = make_uint4(0u, 0u, 0u, 0u), v1 = v0;
      if (avalid) {
        const unsigned short* src = Hbuf + (size_t)(base + am) * FFN + k0 + acol0;
        v0 = *(const uint4*)(src);
        v1 = *(const uint4*)(src + 8);
      }
      *(uint4*)&ldsA[arow * BKP + acol0] = v0;
      *(uint4*)&ldsA[arow * BKP + acol0 + 8] = v1;
    }
#endif
    // ---- fill W2 tile transposed (overlaps the TDM transfer) ----
    {
      const float* s2 = w2base + (size_t)(k0 + brow) * HIDDEN + bcol0;
      __builtin_prefetch(s2 + (size_t)BK * HIDDEN, 0, 1);
#pragma unroll
      for (int j = 0; j < 4; ++j) {
        float4 v = *(const float4*)(s2 + j * 4);
        int n = bcol0 + j * 4;
        ldsB[(n + 0) * BKP + brow] = f32_to_bf16_rne(v.x);
        ldsB[(n + 1) * BKP + brow] = f32_to_bf16_rne(v.y);
        ldsB[(n + 2) * BKP + brow] = f32_to_bf16_rne(v.z);
        ldsB[(n + 3) * BKP + brow] = f32_to_bf16_rne(v.w);
      }
    }
#if USE_TDM
    __builtin_amdgcn_s_wait_tensorcnt(0);  // order TDM LDS writes before barrier
#endif
    __syncthreads();

    v16bf afrag[2];
#pragma unroll
    for (int i = 0; i < 2; ++i) {
      int m = wm * 32 + i * 16 + (lane & 15);
      int kb = (lane >> 4) * 8;
      afrag[i] = frag2(&ldsA[m * BKP + kb], &ldsA[m * BKP + kb + 16]);
    }
#pragma unroll
    for (int j = 0; j < 4; ++j) {
      int n = wn * 64 + j * 16 + (lane & 15);
      int kb = (lane >> 4) * 8;
      v16bf b = frag2(&ldsB[n * BKP + kb], &ldsB[n * BKP + kb + 16]);
#pragma unroll
      for (int i = 0; i < 2; ++i)
        acc[i][j] = __builtin_amdgcn_wmma_f32_16x16x32_bf16(
            false, afrag[i], false, b, (short)0, acc[i][j], false, false);
    }
    __syncthreads();
  }

  // ---- epilogue: scale by combine weight, scatter-add (2 commutative adds/elem) ----
#pragma unroll
  for (int i = 0; i < 2; ++i) {
#pragma unroll
    for (int j = 0; j < 4; ++j) {
      int col = nTile * BN + wn * 64 + j * 16 + (lane & 15);
      int rbase = wm * 32 + i * 16 + (lane >> 4) * 8;
#pragma unroll
      for (int r = 0; r < 8; ++r) {
        int mrow = rbase + r;
        int m = mTile * BM + mrow;
        if (m < cnt) {
          int tok = ldsTok[mrow];
          float w = ldsW[mrow];
          atomicAdd(&Out[(size_t)tok * HIDDEN + col], w * acc[i][j][r]);
        }
      }
    }
  }
}

// ---------------- launch ----------------

extern "C" void kernel_launch(void* const* d_in, const int* in_sizes, int n_in,
                              void* d_out, int out_size, void* d_ws, size_t ws_size,
                              hipStream_t stream) {
  const float* X = (const float*)d_in[0];   // [T, H]
  const float* G = (const float*)d_in[1];   // [H, E]
  const float* W1 = (const float*)d_in[2];  // [E, H, F]
  const float* W3 = (const float*)d_in[3];  // [E, H, F]
  const float* W2 = (const float*)d_in[4];  // [E, F, H]
  float* Out = (float*)d_out;               // [T, H]

  // workspace carve-out
  char* ws = (char*)d_ws;
  size_t off = 0;
  auto carve = [&](size_t bytes) -> void* {
    void* p = ws + off;
    off = (off + bytes + 255) & ~(size_t)255;
    return p;
  };
  int* counts = (int*)carve(NUM_EXPERTS * sizeof(int));
  int* offsets = (int*)carve((NUM_EXPERTS + 1) * sizeof(int));
  int* tokExp = (int*)carve((size_t)TOP_K * TOKENS * sizeof(int));
  int* tokPos = (int*)carve((size_t)TOP_K * TOKENS * sizeof(int));
  float* tokW = (float*)carve((size_t)TOP_K * TOKENS * sizeof(float));
  int* entTok = (int*)carve((size_t)TOP_K * TOKENS * sizeof(int));
  float* entW = (float*)carve((size_t)TOP_K * TOKENS * sizeof(float));
  // +BM rows of slack so padded tile reads never run past the carve.
  unsigned short* Hbuf =
      (unsigned short*)carve((size_t)(TOP_K * TOKENS + BM) * FFN * sizeof(unsigned short));
  (void)ws_size; (void)in_sizes; (void)n_in; (void)out_size;

  // 1) zero output (atomic accumulation target) and counts
  {
    int n4 = TOKENS * HIDDEN / 4;
    zero_f32_kernel<<<(n4 + 255) / 256, 256, 0, stream>>>(Out, n4);
    zero_counts_kernel<<<1, 32, 0, stream>>>(counts);
  }
  // 2) router
  router_kernel<<<TOKENS / 256, 256, 0, stream>>>(X, G, counts, tokExp, tokPos, tokW);
  // 3) prefix sum over 8 experts
  offsets_kernel<<<1, 32, 0, stream>>>(counts, offsets);
  // 4) compact per-expert entry lists
  scatter_kernel<<<(TOP_K * TOKENS + 255) / 256, 256, 0, stream>>>(
      tokExp, tokPos, tokW, offsets, entTok, entW);
  // 5) fused w1/w3 + SwiGLU  (grid covers worst-case M; blocks early-exit on count)
  {
    dim3 grid(FFN / BN, TOKENS / BM, NUM_EXPERTS);
    moe_gemm1_kernel<<<grid, 256, 0, stream>>>(X, W1, W3, counts, offsets, entTok, Hbuf);
  }
  // 6) H @ w2 with combine-weight scatter-add
  {
    dim3 grid(HIDDEN / BN, TOKENS / BM, NUM_EXPERTS);
    moe_gemm2_kernel<<<grid, 256, 0, stream>>>(Hbuf, W2, counts, offsets, entTok, entW, Out);
  }
}
